// AxialTemporalDiscriminator_61366492725894
// MI455X (gfx1250) — matrix operand order repaired and checked by reading, not compiled, not run on hardware
//
#include <hip/hip_runtime.h>

// ---------------------------------------------------------------------------
// AxialTemporalDiscriminator for MI455X (gfx1250, wave32, WMMA)
//
// Phase 1: fused (x + 1e-3*n1 + 1e-4*n2) -> f16 -> [15360x4096]@[4096x8] GEMM
//          via v_wmma_f32_16x16x32_f16. K split x8 (f32 atomic reduction) for
//          ~7680 concurrent waves; W_emb slice staged in LDS once per block so
//          the main streaming loop has NO barriers (per-wave DS ordering).
// Phase 2: per-batch wave: bias+leaky+pos, 2 axial attention blocks,
//          max over T, 24->128->1 MLP, sigmoid.
// ---------------------------------------------------------------------------

typedef __attribute__((ext_vector_type(16))) _Float16 v16h;
typedef __attribute__((ext_vector_type(8)))  _Float16 h8;
typedef __attribute__((ext_vector_type(4)))  _Float16 h4;
typedef __attribute__((ext_vector_type(8)))  float    v8f;
typedef __attribute__((ext_vector_type(4)))  float    f4;

#define ROWS    15360   // B*T*C = 256*20*3
#define KDIM    4096
#define KC      128     // K chunk staged in LDS per iteration
#define KSPLIT  8       // K split for concurrency (each split = 512)
#define KSEG    (KDIM / KSPLIT)   // 512
#define EDIM    8
#define ASTRIDE 136     // halfs per sA row: 272B = 17*16B (aligned, bank-spread)
#define BSTRIDE 520     // halfs per sB row: 1040B = 65*16B (aligned, bank-spread)

union V16H { v16h v; h8 h[2]; };

__global__ __launch_bounds__(256) void fused_embed_wmma(
    const float* __restrict__ x,  const float* __restrict__ n1,
    const float* __restrict__ n2, const float* __restrict__ Wemb,
    float* __restrict__ pbuf)
{
  __shared__ _Float16 sA[8][16][ASTRIDE];   // per-wave A tiles (f16)
  __shared__ _Float16 sB[16][BSTRIDE];      // W_emb K-seg, N-major, N padded to 16

  const int tid  = threadIdx.x;
  const int wave = tid >> 5;
  const int lane = tid & 31;
  const int rowBase = blockIdx.x * 128 + wave * 16;
  const int kbase   = blockIdx.y * KSEG;

  // ---- stage B once: W_emb[kbase..kbase+511][0..7] -> sB[n][k] (f16) ----
  {
    const int n  = tid >> 4;         // 0..15
    const int kg = (tid & 15) * 32;  // 0..480
    if (n < EDIM) {
      #pragma unroll 8
      for (int i = 0; i < 32; ++i)
        sB[n][kg + i] = (_Float16)Wemb[(size_t)(kbase + kg + i) * EDIM + n];
    } else {
      #pragma unroll 8
      for (int i = 0; i < 32; ++i)
        sB[n][kg + i] = (_Float16)0.f;
    }
  }
  __syncthreads();   // the only barrier in this kernel

  v8f acc = {0.f,0.f,0.f,0.f,0.f,0.f,0.f,0.f};

  const int r    = lane & 15;
  const int asel = (lane < 16) ? 0 : 8;    // A frag: lo lanes K+0..7/16..23, hi lanes K+8..15/24..31
  const int bsel = (lane < 16) ? 0 : 8;    // B frag: lo lanes K+0..7/16..23, hi lanes K+8..15/24..31

  for (int kk0 = 0; kk0 < KSEG; kk0 += KC) {
    const int k0 = kbase + kk0;
    // ---- stage A: stream 3 tensors (NT), combine, convert to f16 ----
    #pragma unroll 4
    for (int rr = 0; rr < 16; ++rr) {
      const size_t g = (size_t)(rowBase + rr) * KDIM + k0 + lane * 4;
      f4 xv = __builtin_nontemporal_load((const f4*)(x  + g));
      f4 a1 = __builtin_nontemporal_load((const f4*)(n1 + g));
      f4 a2 = __builtin_nontemporal_load((const f4*)(n2 + g));
      f4 v  = xv + 1.0e-3f * a1 + 1.0e-4f * a2;
      *(h4*)&sA[wave][rr][lane * 4] = __builtin_convertvector(v, h4);
    }
    // ---- prefetch next chunk (global_prefetch_b8) ----
    if (kk0 + KC < KSEG) {
      const size_t gp = (size_t)(rowBase + (lane & 15)) * KDIM + (k0 + KC)
                      + (size_t)(lane >> 4) * 64;
      __builtin_prefetch(x  + gp, 0, 1);
      __builtin_prefetch(n1 + gp, 0, 1);
      __builtin_prefetch(n2 + gp, 0, 1);
    }
    // ---- compute: 4x v_wmma_f32_16x16x32_f16 per chunk ----
    // DS ops are in-order within a wave; sA is wave-private, sB read-only,
    // so no barrier is required between staging and consumption.
    #pragma unroll
    for (int kk = 0; kk < KC; kk += 32) {
      V16H a, b;
      a.h[0] = *(const h8*)&sA[wave][r][kk + asel];
      a.h[1] = *(const h8*)&sA[wave][r][kk + asel + 16];
      b.h[0] = *(const h8*)&sB[r][kk0 + kk + bsel];
      b.h[1] = *(const h8*)&sB[r][kk0 + kk + bsel + 16];
      acc = __builtin_amdgcn_wmma_f32_16x16x32_f16(
          false, a.v, false, b.v, (short)0, acc, false, false);
    }
  }

  // ---- reduce K-splits: D[M=i(+8)][N=lane&15] -> atomic f32 ----
  const int n = lane & 15;
  if (n < EDIM) {
    const int mbase = (lane < 16) ? 0 : 8;
    #pragma unroll
    for (int i = 0; i < 8; ++i) {
      const int row = rowBase + mbase + i;
      atomicAdd(&pbuf[(size_t)row * EDIM + n], acc[i]);
    }
  }
}

// ---------------------------------------------------------------------------
// Phase 2: one wave per batch element.
// ---------------------------------------------------------------------------
__device__ __forceinline__ float leaky02(float v) { return v > 0.f ? v : 0.2f * v; }

__global__ __launch_bounds__(32) void axial_attn_head(
    const float* __restrict__ pbuf, const float* __restrict__ bemb,
    const float* __restrict__ post, const float* __restrict__ posc,
    const float* __restrict__ Wq,   const float* __restrict__ Wkv,
    const float* __restrict__ Wo,   const float* __restrict__ bo,
    const float* __restrict__ Wd1,  const float* __restrict__ bd1,
    const float* __restrict__ Wd2,  const float* __restrict__ bd2,
    float* __restrict__ out)
{
  const int b    = blockIdx.x;
  const int lane = threadIdx.x;

  __shared__ float sh[60][8];       // h tokens: tok = t*3 + c
  __shared__ float sq[60][32];
  __shared__ float sk[60][32];
  __shared__ float sv[60][32];
  __shared__ float so[2][60][32];   // pre-Wo attention outputs (time, channel)
  __shared__ float red[128];

  // load + bias + leaky + axial positional embedding
  for (int i = lane; i < 480; i += 32) {
    const int tok = i >> 3, e = i & 7;
    const int t = tok / 3, c = tok % 3;
    float v = leaky02(pbuf[(size_t)b * 480 + i] + bemb[e]);
    v += post[t * 8 + e] + posc[c * 8 + e];
    sh[tok][e] = v;
  }
  __syncthreads();

  for (int blk = 0; blk < 2; ++blk) {
    for (int axis = 0; axis < 2; ++axis) {
      const float* wq  = Wq  + (size_t)(blk * 2 + axis) * 8 * 32;
      const float* wkv = Wkv + (size_t)(blk * 2 + axis) * 8 * 64;
      // q, k, v projections for all 60 tokens
      for (int tok = lane; tok < 60; tok += 32) {
        float xv[8];
        #pragma unroll
        for (int e = 0; e < 8; ++e) xv[e] = sh[tok][e];
        for (int o = 0; o < 32; ++o) {
          float a = 0.f;
          #pragma unroll
          for (int e = 0; e < 8; ++e) a += xv[e] * wq[e * 32 + o];
          sq[tok][o] = a;
        }
        for (int o = 0; o < 64; ++o) {
          float a = 0.f;
          #pragma unroll
          for (int e = 0; e < 8; ++e) a += xv[e] * wkv[e * 64 + o];
          if (o < 32) sk[tok][o] = a; else sv[tok][o - 32] = a;
        }
      }
      __syncthreads();

      if (axis == 0) {
        // time attention: (c, head) pairs, seq len 20
        if (lane < 24) {
          const int c = lane >> 3, hd = lane & 7;
          for (int i = 0; i < 20; ++i) {
            const int ti = i * 3 + c;
            float d[20], mx = -1e30f;
            for (int j = 0; j < 20; ++j) {
              float s = 0.f;
              #pragma unroll
              for (int e = 0; e < 4; ++e)
                s += sq[ti][hd * 4 + e] * sk[j * 3 + c][hd * 4 + e];
              d[j] = s * 0.5f;               // dh^-0.5, dh=4
              mx = fmaxf(mx, d[j]);
            }
            float den = 0.f;
            for (int j = 0; j < 20; ++j) { d[j] = __expf(d[j] - mx); den += d[j]; }
            const float inv = 1.f / den;
            #pragma unroll
            for (int e = 0; e < 4; ++e) {
              float o = 0.f;
              for (int j = 0; j < 20; ++j) o += d[j] * sv[j * 3 + c][hd * 4 + e];
              so[0][ti][hd * 4 + e] = o * inv;
            }
          }
        }
      } else {
        // channel attention: (t, head) pairs, seq len 3
        for (int p = lane; p < 160; p += 32) {
          const int t = p >> 3, hd = p & 7;
          for (int i = 0; i < 3; ++i) {
            const int ti = t * 3 + i;
            float d[3], mx = -1e30f;
            #pragma unroll
            for (int j = 0; j < 3; ++j) {
              float s = 0.f;
              #pragma unroll
              for (int e = 0; e < 4; ++e)
                s += sq[ti][hd * 4 + e] * sk[t * 3 + j][hd * 4 + e];
              d[j] = s * 0.5f;
              mx = fmaxf(mx, d[j]);
            }
            float den = 0.f;
            #pragma unroll
            for (int j = 0; j < 3; ++j) { d[j] = __expf(d[j] - mx); den += d[j]; }
            const float inv = 1.f / den;
            #pragma unroll
            for (int e = 0; e < 4; ++e) {
              float o = 0.f;
              #pragma unroll
              for (int j = 0; j < 3; ++j) o += d[j] * sv[t * 3 + j][hd * 4 + e];
              so[1][ti][hd * 4 + e] = o * inv;
            }
          }
        }
      }
      __syncthreads();
    }
    // output projections + sum of both axes + leaky
    const float* wo0 = Wo + (size_t)(blk * 2 + 0) * 32 * 8;
    const float* wo1 = Wo + (size_t)(blk * 2 + 1) * 32 * 8;
    const float* b0  = bo + (size_t)(blk * 2 + 0) * 8;
    const float* b1  = bo + (size_t)(blk * 2 + 1) * 8;
    for (int tok = lane; tok < 60; tok += 32) {
      float nh[8];
      #pragma unroll
      for (int e = 0; e < 8; ++e) {
        float s = b0[e] + b1[e];
        for (int o = 0; o < 32; ++o)
          s += so[0][tok][o] * wo0[o * 8 + e] + so[1][tok][o] * wo1[o * 8 + e];
        nh[e] = leaky02(s);
      }
      #pragma unroll
      for (int e = 0; e < 8; ++e) sh[tok][e] = nh[e];
    }
    __syncthreads();
  }

  // max over time -> [C*E] = 24
  if (lane < 24) {
    const int c = lane / 8, e = lane % 8;
    float m = -1e30f;
    for (int t = 0; t < 20; ++t) m = fmaxf(m, sh[t * 3 + c][e]);
    red[lane] = m;
  }
  __syncthreads();

  // dense1 (24->128, leaky) fused with dense2 partial dot
  float part = 0.f;
  for (int o = lane; o < 128; o += 32) {
    float s = bd1[o];
    #pragma unroll
    for (int i = 0; i < 24; ++i) s += red[i] * Wd1[i * 128 + o];
    part += leaky02(s) * Wd2[o];
  }
  __syncthreads();
  red[32 + lane] = part;
  __syncthreads();
  if (lane == 0) {
    float s = bd2[0];
    for (int i = 0; i < 32; ++i) s += red[32 + i];
    out[b] = 1.f / (1.f + __expf(-s));
  }
}

// ---------------------------------------------------------------------------
extern "C" void kernel_launch(void* const* d_in, const int* in_sizes, int n_in,
                              void* d_out, int out_size, void* d_ws, size_t ws_size,
                              hipStream_t stream) {
  (void)in_sizes; (void)n_in; (void)out_size; (void)ws_size;
  const float* x    = (const float*)d_in[0];
  const float* n1   = (const float*)d_in[1];
  const float* n2   = (const float*)d_in[2];
  const float* Wemb = (const float*)d_in[3];
  const float* bemb = (const float*)d_in[4];
  const float* post = (const float*)d_in[5];
  const float* posc = (const float*)d_in[6];
  const float* Wq   = (const float*)d_in[7];
  const float* Wkv  = (const float*)d_in[8];
  const float* Wo   = (const float*)d_in[9];
  const float* bo   = (const float*)d_in[10];
  const float* Wd1  = (const float*)d_in[11];
  const float* bd1  = (const float*)d_in[12];
  const float* Wd2  = (const float*)d_in[13];
  const float* bd2  = (const float*)d_in[14];

  float* pbuf = (float*)d_ws;            // [15360 x 8] f32 partials
  float* outp = (float*)d_out;           // [256] f32

  hipMemsetAsync(pbuf, 0, (size_t)ROWS * EDIM * sizeof(float), stream);

  dim3 g1(ROWS / 128, KSPLIT);           // 120 x 8 blocks, 8 waves each
  fused_embed_wmma<<<g1, 256, 0, stream>>>(x, n1, n2, Wemb, pbuf);

  axial_attn_head<<<256, 32, 0, stream>>>(pbuf, bemb, post, posc,
                                          Wq, Wkv, Wo, bo,
                                          Wd1, bd1, Wd2, bd2, outp);
}
